// MDTA_32427003085217
// MI455X (gfx1250) — matrix operand order repaired and checked by reading, not compiled, not run on hardware
//
#include <hip/hip_runtime.h>

// ---------------------------------------------------------------------------
// MDTA (Restormer transposed channel attention) for gfx1250 / MI455X
// All matmuls via v_wmma_f32_16x16x32_bf16. Internals stored bf16 to halve
// HBM traffic (23.3 TB/s roof); accumulation f32.
// Round 4: TDM A-tile staging, fixed to the 6-arg clang-23 form of
// __builtin_amdgcn_tensor_load_to_lds (extra int32x8 group before cpol).
// ---------------------------------------------------------------------------

#define BSZ   16
#define CCH   384
#define C3    1152
#define HH    64
#define WW    64
#define NPIX  4096          // H*W
#define HEADS 8
#define DHEAD 48            // C / HEADS

typedef __bf16 bf16_t;
typedef __attribute__((ext_vector_type(16))) __bf16       v16bf;
typedef __attribute__((ext_vector_type(8)))  float        v8f;
typedef __attribute__((ext_vector_type(8)))  unsigned int v8u;
typedef __attribute__((ext_vector_type(4)))  unsigned int u32x4;
typedef __attribute__((ext_vector_type(4)))  int          i32x4;
typedef __attribute__((ext_vector_type(8)))  int          i32x8;

union FragU { v8u u; v16bf b; };

__device__ __forceinline__ float bf2f(unsigned short h) {
    unsigned int u = ((unsigned int)h) << 16;
    float f; __builtin_memcpy(&f, &u, 4); return f;
}
__device__ __forceinline__ unsigned short f2bf(float f) {
    unsigned int u; __builtin_memcpy(&u, &f, 4);
    u += 0x7FFFu + ((u >> 16) & 1u);   // round-to-nearest-even
    return (unsigned short)(u >> 16);
}

__device__ __forceinline__ v8f wmma_bf16(v16bf a, v16bf b, v8f c) {
    return __builtin_amdgcn_wmma_f32_16x16x32_bf16(
        false, a, false, b, (short)0, c, false, false);
}

// ---------------------------------------------------------------------------
// CDNA5 TDM: DMA a [rows x cols_halves] bf16 tile (row stride
// row_stride_halves in global memory) into LDS at lds_off, rows packed
// contiguously (row pitch = cols_halves*2 bytes).  D# built per ISA ch.8.
// ---------------------------------------------------------------------------
__device__ __forceinline__ void tdm_load_tile(const unsigned short* gsrc,
                                              unsigned lds_off,
                                              int rows, int cols_halves,
                                              int row_stride_halves,
                                              int tensor_rows) {
    unsigned long long ga = (unsigned long long)(size_t)gsrc;
    u32x4 g0;
    g0[0] = 1u;                                            // count=1, user load
    g0[1] = lds_off;                                       // lds_addr (bytes)
    g0[2] = (unsigned)(ga & 0xFFFFFFFFull);                // global_addr[31:0]
    g0[3] = (unsigned)((ga >> 32) & 0x01FFFFFFull)         // global_addr[56:32]
            | (2u << 30);                                  // type = 2 (image)
    i32x8 g1;
    g1[0] = (int)(1u << 16);                               // data_size=1 (2B), mask=0
    g1[1] = (int)(((unsigned)row_stride_halves & 0xFFFFu) << 16);   // tensor_dim0 lo
    g1[2] = (int)((((unsigned)row_stride_halves) >> 16)
            | (((unsigned)tensor_rows & 0xFFFFu) << 16));  // dim0 hi | tensor_dim1 lo
    g1[3] = (int)((((unsigned)tensor_rows) >> 16)
            | (((unsigned)cols_halves & 0xFFFFu) << 16));  // dim1 hi | tile_dim0
    g1[4] = (int)((unsigned)rows & 0xFFFFu);               // tile_dim1 (tile_dim2=0)
    g1[5] = (int)(unsigned)row_stride_halves;              // tensor_dim0_stride lo32
    g1[6] = 0;                                             // stride hi | dim1_stride
    g1[7] = 0;
    i32x4 z4 = (i32x4)0;                                   // groups 2/3 unused (2D)
    i32x8 z8 = (i32x8)0;                                   // extra group (clang-23)
    __builtin_amdgcn_tensor_load_to_lds(g0, g1, z4, z4, z8, 0);
}

// A fragment (16x32 bf16, row-major source, K contiguous).
// lane L<16: row M=L, K = 0..15 (v0..7); lane>=16: K = 16..31.
__device__ __forceinline__ v16bf load_frag_a(const unsigned int* base,
                                             int stride_u32, int mbase,
                                             int k0_u32, int lane) {
    int row = mbase + (lane & 15);
    int kb  = (lane >> 4) << 3;                 // (lane/16)*16 halves = 8 u32
    const unsigned int* p = base + (size_t)row * stride_u32 + k0_u32 + kb;
    FragU f;
#pragma unroll
    for (int i = 0; i < 4; ++i) f.u[i] = p[i];
#pragma unroll
    for (int i = 0; i < 4; ++i) f.u[4 + i] = p[4 + i];   // +8 halves
    return f.b;
}

// B fragment (32x16 bf16) from an N-major (transposed) tile: tile[n][k],
// K contiguous per n-row.  lane col N = lane%16, K = (lane/16)*16 .. +15.
__device__ __forceinline__ v16bf load_frag_b_t(const unsigned int* base,
                                               int stride_u32, int nbase,
                                               int k0_u32, int lane) {
    int n  = nbase + (lane & 15);
    int kb = (lane >> 4) << 3;                  // (lane/16)*16 halves = 8 u32
    const unsigned int* p = base + (size_t)n * stride_u32 + k0_u32 + kb;
    FragU f;
#pragma unroll
    for (int i = 0; i < 8; ++i) f.u[i] = p[i];
    return f.b;
}

// B fragment from a row-major K x N bf16 matrix (stride ld), zero-padding
// rows k >= kvalid (used for attn@v where K = 48, padded to 64).
__device__ __forceinline__ v16bf load_frag_b_rm(const unsigned short* base,
                                                int ld, int nbase, int kbase,
                                                int kvalid, int lane) {
    int n  = nbase + (lane & 15);
    int kk = kbase + ((lane >> 4) << 4);
    FragU f;
#pragma unroll
    for (int i = 0; i < 8; ++i) {
        int k0 = kk + 2 * i;
        unsigned int lo = (k0     < kvalid) ? base[(size_t)k0       * ld + n] : 0u;
        unsigned int hi = (k0 + 1 < kvalid) ? base[(size_t)(k0 + 1) * ld + n] : 0u;
        f.u[i] = lo | (hi << 16);
    }
    return f.b;
}

// ---------------------------------------------------------------------------
// Kernel 0: convert w1 (1152x384) and w3 (384x384) f32 -> bf16
// ---------------------------------------------------------------------------
__global__ void cvt_weights_kernel(const float* __restrict__ w1,
                                   const float* __restrict__ w3,
                                   unsigned short* __restrict__ w1b,
                                   unsigned short* __restrict__ w3b) {
    int i = blockIdx.x * 256 + threadIdx.x;
    if (i < C3 * CCH)  w1b[i] = f2bf(w1[i]);
    if (i < CCH * CCH) w3b[i] = f2bf(w3[i]);
}

// ---------------------------------------------------------------------------
// Tiled GEMM:  out[b] (M x N) = A (M x K, bf16) @ B[b] (K x N)
// 64x64 block tile, K-step 32, 256 threads = 8 waves (4 M-tiles x 2 N-halves).
// A tile staged by the Tensor Data Mover; B tile transposed into LDS by lanes.
// B_F32:  B source is f32 (converted to bf16 while staging into LDS)
// OUT_F32: output f32, else bf16.
// ---------------------------------------------------------------------------
template <bool B_F32, bool OUT_F32>
__global__ void gemm_bf16_kernel(const unsigned short* __restrict__ A,
                                 const void* __restrict__ Bsrc,
                                 void* __restrict__ Out,
                                 int M, int N, int K,
                                 size_t bStride, size_t outStride) {
    __shared__ alignas(16) unsigned short shA[64 * 32];  // [m][k]
    __shared__ alignas(16) unsigned short shB[64 * 32];  // [n][k]  (transposed)

    const int tid  = threadIdx.x;
    const int wave = tid >> 5;
    const int lane = tid & 31;
    const int n0   = blockIdx.x * 64;
    const int m0   = blockIdx.y * 64;
    const int z    = blockIdx.z;

    v8f acc[2];
#pragma unroll
    for (int t = 0; t < 2; ++t) acc[t] = (v8f){};

#pragma unroll 1
    for (int k0 = 0; k0 < K; k0 += 32) {
        // ---- stage A tile (64 rows x 32 halves) via TDM DMA ----
        if (wave == 0)
            tdm_load_tile(A + (size_t)m0 * K + k0,
                          (unsigned)(size_t)&shA[0],
                          /*rows=*/64, /*cols=*/32,
                          /*row_stride=*/K, /*tensor_rows=*/M);

        // ---- stage B tile transposed (k in [k0,k0+32) x n in [n0,n0+64)) ----
        if (B_F32) {
            const float* Bp = (const float*)Bsrc + (size_t)z * bStride;
#pragma unroll
            for (int j = 0; j < 8; ++j) {
                int idx = tid + j * 256;
                int kk = idx >> 6, nn = idx & 63;
                shB[nn * 32 + kk] = f2bf(Bp[(size_t)(k0 + kk) * N + (n0 + nn)]);
            }
            if (k0 + 32 < K)
                __builtin_prefetch(Bp + (size_t)(k0 + 32 + (tid >> 6)) * N + n0 + (tid & 63), 0, 0);
        } else {
            const unsigned short* Bp = (const unsigned short*)Bsrc + (size_t)z * bStride;
#pragma unroll
            for (int j = 0; j < 8; ++j) {
                int idx = tid + j * 256;
                int kk = idx >> 6, nn = idx & 63;
                shB[nn * 32 + kk] = Bp[(size_t)(k0 + kk) * N + (n0 + nn)];
            }
            if (k0 + 32 < K)
                __builtin_prefetch(Bp + (size_t)(k0 + 32 + (tid >> 6)) * N + n0 + (tid & 63), 0, 0);
        }

        if (wave == 0)
            __builtin_amdgcn_s_wait_tensorcnt(0);   // TDM tile landed in LDS
        __syncthreads();

        // ---- compute: each wave = one 16-row M tile x 32 N columns ----
        v16bf af = load_frag_a((const unsigned int*)shA, 16, (wave & 3) * 16, 0, lane);
#pragma unroll
        for (int t = 0; t < 2; ++t) {
            v16bf bf = load_frag_b_t((const unsigned int*)shB, 16,
                                     (wave >> 2) * 32 + t * 16, 0, lane);
            acc[t] = wmma_bf16(af, bf, acc[t]);
        }
        __syncthreads();
    }

    // ---- write out ----
#pragma unroll
    for (int t = 0; t < 2; ++t) {
        int ncol  = n0 + (wave >> 2) * 32 + t * 16 + (lane & 15);
        int mrow0 = m0 + (wave & 3) * 16 + ((lane >> 4) << 3);
        if (OUT_F32) {
            float* O = (float*)Out + (size_t)z * outStride;
#pragma unroll
            for (int i = 0; i < 8; ++i)
                O[(size_t)(mrow0 + i) * N + ncol] = acc[t][i];
        } else {
            unsigned short* O = (unsigned short*)Out + (size_t)z * outStride;
#pragma unroll
            for (int i = 0; i < 8; ++i)
                O[(size_t)(mrow0 + i) * N + ncol] = f2bf(acc[t][i]);
        }
    }
}

// ---------------------------------------------------------------------------
// Kernel: 3x3 depthwise conv, SAME zero padding, bf16 in/out, f32 accumulate
// ---------------------------------------------------------------------------
__global__ void dwconv3x3_kernel(const unsigned short* __restrict__ in,
                                 const float* __restrict__ w2,
                                 unsigned short* __restrict__ out) {
    long long idx = (long long)blockIdx.x * 256 + threadIdx.x;
    int x  = (int)(idx & 63);
    int y  = (int)((idx >> 6) & 63);
    int ch = (int)((idx >> 12) % C3);
    int b  = (int)(idx / ((long long)C3 * NPIX));
    const unsigned short* base = in + ((size_t)b * C3 + ch) * NPIX;
    const float* wp = w2 + ch * 9;
    float acc = 0.f;
#pragma unroll
    for (int dy = -1; dy <= 1; ++dy) {
        int yy = y + dy;
        if (yy < 0 || yy >= HH) continue;
#pragma unroll
        for (int dx = -1; dx <= 1; ++dx) {
            int xx = x + dx;
            if (xx < 0 || xx >= WW) continue;
            acc += wp[(dy + 1) * 3 + (dx + 1)] * bf2f(base[yy * WW + xx]);
        }
    }
    out[idx] = f2bf(acc);
}

// ---------------------------------------------------------------------------
// Kernel: per-(b,head) transposed attention.
//   q,k,v : 48 x 4096 bf16 rows of dw buffer
//   G = q @ k^T (WMMA, K split over 8 waves, LDS f32 atomic reduce)
//   softmax(G * rq_i * rk_j * temperature)  -> bf16 attn in LDS (48x64 padded)
//   out = attn @ v (WMMA, K=48 padded to 64) -> bf16
// ---------------------------------------------------------------------------
__global__ void attn_kernel(const unsigned short* __restrict__ dw,
                            const float* __restrict__ scaling,
                            unsigned short* __restrict__ attn_out) {
    __shared__ float shG[DHEAD * DHEAD];
    __shared__ float shRn[2 * DHEAD];
    __shared__ alignas(16) unsigned short shAttn[DHEAD * 64];

    const int bh = blockIdx.x;
    const int b  = bh >> 3;
    const int h  = bh & 7;
    const int tid  = threadIdx.x;
    const int wave = tid >> 5;
    const int lane = tid & 31;

    const unsigned short* qp = dw + ((size_t)b * C3 +           h * DHEAD) * NPIX;
    const unsigned short* kp = dw + ((size_t)b * C3 + CCH     + h * DHEAD) * NPIX;
    const unsigned short* vp = dw + ((size_t)b * C3 + 2 * CCH + h * DHEAD) * NPIX;

    for (int i = tid; i < DHEAD * DHEAD; i += 256) shG[i] = 0.f;
    for (int i = tid; i < DHEAD * 64;    i += 256) shAttn[i] = 0;

    // ---- phase 1: 1/max(||row||,1e-12) for 48 q rows + 48 k rows ----
#pragma unroll 1
    for (int r = wave * 12; r < wave * 12 + 12; ++r) {
        const unsigned short* rp = (r < DHEAD) ? qp + (size_t)r * NPIX
                                               : kp + (size_t)(r - DHEAD) * NPIX;
        float s = 0.f;
        for (int j = lane; j < NPIX; j += 32) {
            float f = bf2f(rp[j]);
            s += f * f;
        }
#pragma unroll
        for (int off = 16; off; off >>= 1) s += __shfl_xor(s, off);
        if (lane == 0) shRn[r] = 1.0f / fmaxf(sqrtf(s), 1e-12f);
    }
    __syncthreads();

    // ---- phase 2: G = q @ k^T, each wave handles 512 of the K=4096 dim ----
    // unroll 1 keeps live state at 9 accum frags + 6 operand frags -> no spills
    v8f acc[3][3];
#pragma unroll
    for (int mt = 0; mt < 3; ++mt)
#pragma unroll
        for (int nt = 0; nt < 3; ++nt) acc[mt][nt] = (v8f){};

    const unsigned int* qu = (const unsigned int*)qp;
    const unsigned int* ku = (const unsigned int*)kp;
    const int kbeg = wave * 512;
#pragma unroll 1
    for (int kk = kbeg; kk < kbeg + 512; kk += 32) {
        v16bf af[3], bfr[3];
#pragma unroll
        for (int mt = 0; mt < 3; ++mt)
            af[mt] = load_frag_a(qu, NPIX / 2, mt * 16, kk >> 1, lane);
#pragma unroll
        for (int nt = 0; nt < 3; ++nt)
            bfr[nt] = load_frag_b_t(ku, NPIX / 2, nt * 16, kk >> 1, lane);
#pragma unroll
        for (int mt = 0; mt < 3; ++mt)
#pragma unroll
            for (int nt = 0; nt < 3; ++nt)
                acc[mt][nt] = wmma_bf16(af[mt], bfr[nt], acc[mt][nt]);
    }
#pragma unroll
    for (int mt = 0; mt < 3; ++mt)
#pragma unroll
        for (int nt = 0; nt < 3; ++nt) {
            int ncol = nt * 16 + (lane & 15);
            int mr   = mt * 16 + ((lane >> 4) << 3);
#pragma unroll
            for (int i = 0; i < 8; ++i)
                atomicAdd(&shG[(mr + i) * DHEAD + ncol], acc[mt][nt][i]);
        }
    __syncthreads();

    // ---- phase 3: scale + softmax rows (48 threads, one row each) ----
    if (tid < DHEAD) {
        const float sc = scaling[h];
        const float rq = shRn[tid];
        float mx = -1e30f;
        for (int j = 0; j < DHEAD; ++j) {
            float s = shG[tid * DHEAD + j] * rq * shRn[DHEAD + j] * sc;
            shG[tid * DHEAD + j] = s;
            mx = fmaxf(mx, s);
        }
        float sum = 0.f;
        for (int j = 0; j < DHEAD; ++j) {
            float e = __expf(shG[tid * DHEAD + j] - mx);
            shG[tid * DHEAD + j] = e;
            sum += e;
        }
        float inv = 1.0f / sum;
        for (int j = 0; j < DHEAD; ++j)
            shAttn[tid * 64 + j] = f2bf(shG[tid * DHEAD + j] * inv);
    }
    __syncthreads();

    // ---- phase 4: out = attn @ v  (M=48, K=48 padded to 64, N=4096) ----
    const unsigned int* au = (const unsigned int*)shAttn;
#pragma unroll 1
    for (int g = 0; g < 32; ++g) {
        const int nb = wave * 512 + g * 16;
        v8f o[3];
#pragma unroll
        for (int mt = 0; mt < 3; ++mt) o[mt] = (v8f){};
#pragma unroll
        for (int s = 0; s < 2; ++s) {
            v16bf bfv = load_frag_b_rm(vp, NPIX, nb, s * 32, DHEAD, lane);
#pragma unroll
            for (int mt = 0; mt < 3; ++mt) {
                v16bf af = load_frag_a(au, 32, mt * 16, s * 16, lane);
                o[mt] = wmma_bf16(af, bfv, o[mt]);
            }
        }
        const int ncol = nb + (lane & 15);
#pragma unroll
        for (int mt = 0; mt < 3; ++mt) {
            int ch0 = h * DHEAD + mt * 16 + ((lane >> 4) << 3);
#pragma unroll
            for (int i = 0; i < 8; ++i)
                attn_out[((size_t)b * CCH + ch0 + i) * NPIX + ncol] = f2bf(o[mt][i]);
        }
    }
}

// ---------------------------------------------------------------------------
// Launch
// ---------------------------------------------------------------------------
extern "C" void kernel_launch(void* const* d_in, const int* in_sizes, int n_in,
                              void* d_out, int out_size, void* d_ws, size_t ws_size,
                              hipStream_t stream) {
    const float* x       = (const float*)d_in[0];
    const float* w1      = (const float*)d_in[1];
    const float* w2      = (const float*)d_in[2];
    const float* w3      = (const float*)d_in[3];
    const float* scaling = (const float*)d_in[4];
    float* out = (float*)d_out;

    // workspace layout (bytes)
    const size_t QKV_BYTES = (size_t)BSZ * C3 * NPIX * 2;   // 150,994,944
    const size_t AO_BYTES  = (size_t)BSZ * CCH * NPIX * 2;  //  50,331,648
    char* ws = (char*)d_ws;
    unsigned short* qkv  = (unsigned short*)(ws);
    unsigned short* dwb  = (unsigned short*)(ws + QKV_BYTES);
    unsigned short* aout = (unsigned short*)(ws + 2 * QKV_BYTES);
    unsigned short* w1b  = (unsigned short*)(ws + 2 * QKV_BYTES + AO_BYTES);
    unsigned short* w3b  = (unsigned short*)(ws + 2 * QKV_BYTES + AO_BYTES
                                                + (size_t)C3 * CCH * 2);

    // 0) weights -> bf16
    cvt_weights_kernel<<<(C3 * CCH + 255) / 256, 256, 0, stream>>>(w1, w3, w1b, w3b);

    // 1) conv1: qkv[b] = w1 @ x[b]   (M=1152, N=4096, K=384)
    gemm_bf16_kernel<true, false><<<dim3(NPIX / 64, C3 / 64, BSZ), 256, 0, stream>>>(
        w1b, x, qkv, C3, NPIX, CCH,
        (size_t)CCH * NPIX, (size_t)C3 * NPIX);

    // 2) 3x3 depthwise
    dwconv3x3_kernel<<<(BSZ * C3 * NPIX) / 256, 256, 0, stream>>>(qkv, w2, dwb);

    // 3) attention per (b, head)
    attn_kernel<<<BSZ * HEADS, 256, 0, stream>>>(dwb, scaling, aout);

    // 4) conv3: out[b] = w3 @ attn_out[b]   (M=384, N=4096, K=384), f32 out
    gemm_bf16_kernel<false, true><<<dim3(NPIX / 64, CCH / 64, BSZ), 256, 0, stream>>>(
        w3b, aout, out, CCH, NPIX, CCH,
        (size_t)CCH * NPIX, (size_t)CCH * NPIX);
}
